// So3krates_5153960755282
// MI455X (gfx1250) — matrix operand order repaired and checked by reading, not compiled, not run on hardware
//
#include <hip/hip_runtime.h>
#include <math.h>

typedef unsigned short ushort_t;
typedef __attribute__((ext_vector_type(16))) __bf16 v16bf;
typedef __attribute__((ext_vector_type(8)))  float  v8f;
typedef __attribute__((ext_vector_type(4)))  unsigned int u32x4;

#define INV_AVG 0.0625f
#define R_MAXF  5.0f
#define PI_F    3.14159265358979323846f
#define RSQRT_DH 0.17677669529663687f   /* 1/sqrt(32) */

// ---------------- bf16 helpers ----------------
__device__ __forceinline__ ushort_t f2bf(float f) {
    unsigned int u = __float_as_uint(f);
    return (ushort_t)((u + 0x7FFFu + ((u >> 16) & 1u)) >> 16);   // RNE
}
__device__ __forceinline__ float bf2f(ushort_t h) {
    return __uint_as_float(((unsigned int)h) << 16);
}
// fast silu: x * v_rcp_f32(1+exp(-x)) -- avoids IEEE div_scale/div_fmas chain
__device__ __forceinline__ float silu_f(float x) {
    return x * __builtin_amdgcn_rcpf(1.0f + __expf(-x));
}
__device__ __forceinline__ float wave_sum(float v) {
#pragma unroll
    for (int m = 16; m > 0; m >>= 1) v += __shfl_xor(v, m, 32);
    return v;
}

// ---------------- WMMA fragment loads (wave32, 16x16x32 bf16) ----------------
// A 16x32 layout: lanes 0-15 hold M=lane, K = {kb..kb+7} (VGPR0-3) and {kb+16..kb+23} (VGPR4-7);
// lanes 16-31 hold same M with K offset +8.  Both halves are two contiguous 16B chunks.
union FragU { v16bf v; u32x4 q[2]; };

__device__ __forceinline__ v16bf load_a_frag(const ushort_t* A, int lda, int row, int kb, int hl) {
    FragU f;
    const ushort_t* p = A + (size_t)row * lda + kb + hl * 8;
    f.q[0] = *(const u32x4*)p;
    f.q[1] = *(const u32x4*)(p + 16);
    return f.v;
}
// B 32x16 from column-major-transposed weights Bt[N][Kp]: lane holds column n=lane&15,
// K = {kb+hl*8..+7} and {kb+16+hl*8..+7} (same order as A).
__device__ __forceinline__ v16bf load_b_frag(const ushort_t* Bt, int Kp, int ncol, int kb, int hl) {
    FragU f;
    const ushort_t* p = Bt + (size_t)ncol * Kp + kb + hl * 8;
    f.q[0] = *(const u32x4*)p;
    f.q[1] = *(const u32x4*)(p + 16);
    return f.v;
}

// ---------------- generic WMMA GEMM: C[M x Nt*16] = A[M x Kp] @ B ----------------
// B panel (16 cols x Kp) is staged once per block into LDS with async-to-LDS copies
// (all 8 waves of the block use the same ntile), then read back as ds_load_b128.
template<bool BETA, bool SILU, bool OUTBF>
__global__ __launch_bounds__(256)
void gemm_bf16(const ushort_t* __restrict__ A, int M, int Kp,
               const ushort_t* __restrict__ Bt,
               const float* __restrict__ bias, int biasLen,
               float* __restrict__ Cf, ushort_t* __restrict__ Cb, int ldc)
{
    __shared__ ushort_t shB[16 * 160];     // up to Kp=160
    int tid  = threadIdx.x;
    int lane = tid & 31;
    int wv   = tid >> 5;
    int ntile = blockIdx.y;

    // cooperative async stage of the B panel: 16B per transfer, ASYNCcnt-tracked
    {
        const ushort_t* src = Bt + (size_t)ntile * 16 * Kp;
        int chunks = (16 * Kp) >> 3;       // number of 16-byte chunks
        for (int c = tid; c < chunks; c += 256) {
            unsigned dstOff = (unsigned)(size_t)(&shB[c * 8]);
            const ushort_t* gp = src + c * 8;
            asm volatile("global_load_async_to_lds_b128 %0, %1, off"
                         :: "v"(dstOff), "v"(gp) : "memory");
        }
        asm volatile("s_wait_asynccnt 0" ::: "memory");
    }
    __syncthreads();

    int mtile = blockIdx.x * 8 + wv;
    bool active = (mtile < ((M + 15) >> 4));   // wave-uniform
    if (active) {
        int hl = lane >> 4;
        int nn = lane & 15;
        int col = ntile * 16 + nn;
        int arow = mtile * 16 + nn;
        if (arow >= M) arow = M - 1;

        v8f acc;
#pragma unroll
        for (int i = 0; i < 8; i++) acc[i] = 0.0f;
        if (BETA) {
#pragma unroll
            for (int i = 0; i < 8; i++) {
                int m = mtile * 16 + i + hl * 8;
                acc[i] = (m < M) ? Cf[(size_t)m * ldc + col] : 0.0f;
            }
        }
        for (int kb = 0; kb < Kp; kb += 32) {
            if (kb + 32 < Kp) __builtin_prefetch(A + (size_t)arow * Kp + kb + 32, 0, 0);
            v16bf a = load_a_frag(A, Kp, arow, kb, hl);
            v16bf b = load_b_frag(shB, Kp, nn, kb, hl);
            acc = __builtin_amdgcn_wmma_f32_16x16x32_bf16(false, a, false, b, (short)0, acc, false, false);
        }
        float bv = 0.0f;
        if (bias != nullptr && col < biasLen) bv = bias[col];
#pragma unroll
        for (int i = 0; i < 8; i++) {
            int m = mtile * 16 + i + hl * 8;
            if (m >= M) continue;
            float x = acc[i] + bv;
            if (SILU) x = silu_f(x);
            if (OUTBF) Cb[(size_t)m * ldc + col] = f2bf(x);
            else       Cf[(size_t)m * ldc + col] = x;
        }
    }
}

// ---------------- fused edge radial filter: w = silu(R@W1+b1)@W2 + dd@Wdeg ----------------
__global__ __launch_bounds__(256)
void edge_filter(const ushort_t* __restrict__ RBFb,   // [E][32] bf16
                 const float* __restrict__ ev,        // [N][16]
                 const int* __restrict__ snd, const int* __restrict__ rcv,
                 const ushort_t* __restrict__ W1t,    // [128][32] bf16 col-major
                 const float* __restrict__ b1,        // [128]
                 const ushort_t* __restrict__ W2t,    // [128][128] bf16 col-major
                 const float* __restrict__ Wdeg,      // [4][128]
                 ushort_t* __restrict__ Wout,         // [E][128] bf16
                 int E)
{
    __shared__ ushort_t hid[8][16][128];   // 32 KB
    __shared__ float    ddl[8][16][4];
    int lane = threadIdx.x & 31, wv = threadIdx.x >> 5;
    int etile = blockIdx.x * 8 + wv;
    bool active = (etile * 16 < E);
    int hl = lane >> 4, nn = lane & 15;

    if (active && lane < 16) {             // per-edge degree contraction dd
        int e = etile * 16 + lane;
        int s = snd[e], r = rcv[e];
        const float* es = ev + (size_t)s * 16;
        const float* er = ev + (size_t)r * 16;
        float d0 = es[0] * er[0], d1 = 0.f, d2 = 0.f, d3 = 0.f;
        for (int c = 1; c < 4;  c++) d1 += es[c] * er[c];
        for (int c = 4; c < 9;  c++) d2 += es[c] * er[c];
        for (int c = 9; c < 16; c++) d3 += es[c] * er[c];
        ddl[wv][lane][0] = d0; ddl[wv][lane][1] = d1;
        ddl[wv][lane][2] = d2; ddl[wv][lane][3] = d3;
    }
    if (active) {                          // layer 1: [16x32] @ [32x128] + silu -> LDS bf16
        int arow = etile * 16 + nn;
        v16bf a = load_a_frag(RBFb, 32, arow, 0, hl);
        for (int nt = 0; nt < 8; nt++) {
            v16bf b = load_b_frag(W1t, 32, nt * 16 + nn, 0, hl);
            v8f acc;
#pragma unroll
            for (int i = 0; i < 8; i++) acc[i] = 0.0f;
            acc = __builtin_amdgcn_wmma_f32_16x16x32_bf16(false, a, false, b, (short)0, acc, false, false);
            int colc = nt * 16 + nn;
            float bb = b1[colc];
#pragma unroll
            for (int i = 0; i < 8; i++) {
                int m = i + hl * 8;
                hid[wv][m][colc] = f2bf(silu_f(acc[i] + bb));
            }
        }
    }
    __syncthreads();
    if (active) {                          // layer 2: [16x128] @ [128x128], A from LDS
        const ushort_t* hp = &hid[wv][0][0];
        v16bf a0 = load_a_frag(hp, 128, nn, 0,  hl);
        v16bf a1 = load_a_frag(hp, 128, nn, 32, hl);
        v16bf a2 = load_a_frag(hp, 128, nn, 64, hl);
        v16bf a3 = load_a_frag(hp, 128, nn, 96, hl);
        for (int nt = 0; nt < 8; nt++) {
            int colc = nt * 16 + nn;
            v8f acc;
#pragma unroll
            for (int i = 0; i < 8; i++) acc[i] = 0.0f;
            acc = __builtin_amdgcn_wmma_f32_16x16x32_bf16(false, a0, false, load_b_frag(W2t, 128, colc, 0,  hl), (short)0, acc, false, false);
            acc = __builtin_amdgcn_wmma_f32_16x16x32_bf16(false, a1, false, load_b_frag(W2t, 128, colc, 32, hl), (short)0, acc, false, false);
            acc = __builtin_amdgcn_wmma_f32_16x16x32_bf16(false, a2, false, load_b_frag(W2t, 128, colc, 64, hl), (short)0, acc, false, false);
            acc = __builtin_amdgcn_wmma_f32_16x16x32_bf16(false, a3, false, load_b_frag(W2t, 128, colc, 96, hl), (short)0, acc, false, false);
            float w0 = Wdeg[colc], w1 = Wdeg[128 + colc], w2 = Wdeg[256 + colc], w3 = Wdeg[384 + colc];
#pragma unroll
            for (int i = 0; i < 8; i++) {
                int m = i + hl * 8;
                float x = acc[i] + ddl[wv][m][0] * w0 + ddl[wv][m][1] * w1
                                 + ddl[wv][m][2] * w2 + ddl[wv][m][3] * w3;
                Wout[(size_t)(etile * 16 + m) * 128 + colc] = f2bf(x);
            }
        }
    }
}

// ---------------- edge geometry: Y, cutoff, RBF, ev seed ----------------
__global__ __launch_bounds__(256)
void edge_geom(const float* __restrict__ pos, const int* __restrict__ snd,
               const int* __restrict__ rcv, float* __restrict__ Y,
               float* __restrict__ Cc, ushort_t* __restrict__ RBFb,
               float* __restrict__ ev, int E)
{
    int e = blockIdx.x * blockDim.x + threadIdx.x;
    if (e >= E) return;
    int s = snd[e], r = rcv[e];
    float vx = pos[r * 3 + 0] - pos[s * 3 + 0];
    float vy = pos[r * 3 + 1] - pos[s * 3 + 1];
    float vz = pos[r * 3 + 2] - pos[s * 3 + 2];
    float rn = sqrtf(vx * vx + vy * vy + vz * vz);
    float inv = __builtin_amdgcn_rcpf(fmaxf(rn, 1e-9f));
    float x = vx * inv, y = vy * inv, z = vz * inv;
    float x2 = x * x, y2 = y * y, z2 = z * z;
    const float s3 = 1.7320508075688772f, s5 = 2.23606797749979f, s15 = 3.872983346207417f;
    const float s70 = 8.366600265340756f, s105 = 10.246950765959598f, s42 = 6.48074069840786f, s7 = 2.6457513110645907f;
    float Yv[16];
    Yv[0] = 1.0f;
    Yv[1] = s3 * x; Yv[2] = s3 * y; Yv[3] = s3 * z;
    Yv[4] = s15 * x * y; Yv[5] = s15 * y * z; Yv[6] = 0.5f * s5 * (3.0f * z2 - 1.0f);
    Yv[7] = s15 * x * z; Yv[8] = 0.5f * s15 * (x2 - y2);
    Yv[9]  = 0.25f * s70 * y * (3.0f * x2 - y2);
    Yv[10] = s105 * x * y * z;
    Yv[11] = 0.25f * s42 * y * (5.0f * z2 - 1.0f);
    Yv[12] = 0.5f * s7 * z * (5.0f * z2 - 3.0f);
    Yv[13] = 0.25f * s42 * x * (5.0f * z2 - 1.0f);
    Yv[14] = 0.5f * s105 * z * (x2 - y2);
    Yv[15] = 0.25f * s70 * x * (x2 - 3.0f * y2);
    float cut = (rn < R_MAXF) ? 0.5f * (__cosf(PI_F * rn / R_MAXF) + 1.0f) : 0.0f;
    Cc[e] = cut;
#pragma unroll
    for (int c = 0; c < 16; c++) {
        Y[(size_t)e * 16 + c] = Yv[c];
        atomicAdd(&ev[(size_t)r * 16 + c], Yv[c] * cut);
    }
    const float step = R_MAXF / 31.0f, gamma = 40.96f; // (32/5)^2
#pragma unroll
    for (int j = 0; j < 32; j++) {
        float d = rn - (float)j * step;
        RBFb[(size_t)e * 32 + j] = f2bf(__expf(-gamma * d * d));
    }
}

// ---------------- invariant attention: alpha + scatter of alpha*v ----------------
__global__ __launch_bounds__(256)
void attn_inv(const float* __restrict__ q, const float* __restrict__ k,
              const float* __restrict__ v, const ushort_t* __restrict__ wbf,
              const float* __restrict__ Cc, const int* __restrict__ snd,
              const int* __restrict__ rcv, float* __restrict__ agg, int E)
{
    int lane = threadIdx.x & 31;
    int e = (int)((blockIdx.x * 256u + threadIdx.x) >> 5);
    if (e >= E) return;
    int s = snd[e], r = rcv[e];
    float cc = Cc[e];
    size_t qb = (size_t)r * 128, kb = (size_t)s * 128, eb = (size_t)e * 128;
#pragma unroll
    for (int h = 0; h < 4; h++) {
        int idx = h * 32 + lane;
        float p = q[qb + idx] * bf2f(wbf[eb + idx]) * k[kb + idx];
        float alpha = wave_sum(p) * RSQRT_DH * cc;
        atomicAdd(&agg[qb + idx], alpha * v[kb + idx]);
    }
}

// ---------------- geometric attention: a2 per degree, scatter into ev ----------------
__global__ __launch_bounds__(256)
void attn_ev(const float* __restrict__ q2, const float* __restrict__ k2,
             const ushort_t* __restrict__ wbf, const float* __restrict__ Cc,
             const float* __restrict__ Y, const int* __restrict__ snd,
             const int* __restrict__ rcv, float* __restrict__ ev, int E)
{
    int lane = threadIdx.x & 31;
    int e = (int)((blockIdx.x * 256u + threadIdx.x) >> 5);
    if (e >= E) return;
    int s = snd[e], r = rcv[e];
    float cc = Cc[e];
    size_t qb = (size_t)r * 128, kb = (size_t)s * 128, eb = (size_t)e * 128;
    float a0, a1, a2, a3;
    { float p = q2[qb + lane]      * bf2f(wbf[eb + lane])      * k2[kb + lane];      a0 = wave_sum(p) * RSQRT_DH * cc; }
    { float p = q2[qb + 32 + lane] * bf2f(wbf[eb + 32 + lane]) * k2[kb + 32 + lane]; a1 = wave_sum(p) * RSQRT_DH * cc; }
    { float p = q2[qb + 64 + lane] * bf2f(wbf[eb + 64 + lane]) * k2[kb + 64 + lane]; a2 = wave_sum(p) * RSQRT_DH * cc; }
    { float p = q2[qb + 96 + lane] * bf2f(wbf[eb + 96 + lane]) * k2[kb + 96 + lane]; a3 = wave_sum(p) * RSQRT_DH * cc; }
    if (lane < 16) {
        int dg = (lane == 0) ? 0 : (lane < 4) ? 1 : (lane < 9) ? 2 : 3;
        float av = (dg == 0) ? a0 : (dg == 1) ? a1 : (dg == 2) ? a2 : a3;
        atomicAdd(&ev[(size_t)r * 16 + lane], av * Y[(size_t)e * 16 + lane] * INV_AVG);
    }
}

// ---------------- small elementwise kernels ----------------
__global__ void conv_wT(const float* __restrict__ src, int K, int Nc,
                        ushort_t* __restrict__ dst, int Kp, int Np)
{
    int i = blockIdx.x * blockDim.x + threadIdx.x;
    if (i >= Np * Kp) return;
    int n = i / Kp, k = i % Kp;
    float v = (n < Nc && k < K) ? src[(size_t)k * Nc + n] : 0.0f;
    dst[i] = f2bf(v);
}
__global__ void conv_scale(const float* __restrict__ src, ushort_t* __restrict__ dst,
                           int n, float scale)
{
    int i = blockIdx.x * blockDim.x + threadIdx.x;
    if (i < n) dst[i] = f2bf(src[i] * scale);
}
__global__ void scale_buf(float* __restrict__ b, int n, float s)
{
    int i = blockIdx.x * blockDim.x + threadIdx.x;
    if (i < n) b[i] *= s;
}
__global__ void f_init(float* __restrict__ f, const float* __restrict__ embed,
                       const int* __restrict__ species, int N)
{
    int i = blockIdx.x * blockDim.x + threadIdx.x;
    if (i >= N * 128) return;
    f[i] = embed[(size_t)species[i >> 7] * 128 + (i & 127)];
}
__global__ void build_h(const float* __restrict__ f, const float* __restrict__ ev,
                        ushort_t* __restrict__ hb, int N)
{
    int i = blockIdx.x * blockDim.x + threadIdx.x;
    if (i >= N * 160) return;
    int n = i / 160, c = i % 160;
    float x = 0.0f;
    if (c < 128) x = f[(size_t)n * 128 + c];
    else if (c < 132) {
        int l = c - 128;
        int o0 = (l == 0) ? 0 : (l == 1) ? 1 : (l == 2) ? 4 : 9;
        int o1 = (l == 0) ? 1 : (l == 1) ? 4 : (l == 2) ? 9 : 16;
        const float* e = ev + (size_t)n * 16;
        float s = 0.0f;
        for (int j = o0; j < o1; j++) s += e[j] * e[j];
        x = s;
    }
    hb[i] = f2bf(x);
}
__global__ void upd_f(float* __restrict__ f, const float* __restrict__ yb, int N)
{
    int i = blockIdx.x * blockDim.x + threadIdx.x;
    if (i >= N * 128) return;
    int n = i >> 7, c = i & 127;
    f[i] += yb[(size_t)n * 144 + c];
}
__global__ void upd_ev(float* __restrict__ ev, const float* __restrict__ yb, int N)
{
    int i = blockIdx.x * blockDim.x + threadIdx.x;
    if (i >= N * 16) return;
    int n = i >> 4, c = i & 15;
    int dg = (c == 0) ? 0 : (c < 4) ? 1 : (c < 9) ? 2 : 3;
    ev[i] *= (1.0f + yb[(size_t)n * 144 + 128 + dg]);
}
__global__ __launch_bounds__(256)
void out_energy(const float* __restrict__ hid, const float* __restrict__ Wo2,
                const float* __restrict__ bo2, const int* __restrict__ batch,
                float* __restrict__ out, int N)
{
    int lane = threadIdx.x & 31;
    int n = (int)((blockIdx.x * 256u + threadIdx.x) >> 5);
    if (n >= N) return;
    float sum = 0.0f;
#pragma unroll
    for (int j = 0; j < 4; j++) sum += hid[(size_t)n * 128 + j * 32 + lane] * Wo2[j * 32 + lane];
    sum = wave_sum(sum);
    if (lane == 0) atomicAdd(&out[batch[n]], sum + bo2[0]);
}

// ---------------- host launcher ----------------
extern "C" void kernel_launch(void* const* d_in, const int* in_sizes, int n_in,
                              void* d_out, int out_size, void* d_ws, size_t ws_size,
                              hipStream_t stream)
{
    (void)n_in; (void)ws_size;
    const float* positions = (const float*)d_in[0];
    const int*   species   = (const int*)d_in[1];
    const int*   senders   = (const int*)d_in[2];
    const int*   receivers = (const int*)d_in[3];
    const int*   batch     = (const int*)d_in[4];
    const float* embed     = (const float*)d_in[5];
    const float* Wq   = (const float*)d_in[6];
    const float* Wk   = (const float*)d_in[7];
    const float* Wv   = (const float*)d_in[8];
    const float* Wo   = (const float*)d_in[9];
    const float* Wrbf1 = (const float*)d_in[10];
    const float* brbf1 = (const float*)d_in[11];
    const float* Wrbf2 = (const float*)d_in[12];
    const float* Wdeg  = (const float*)d_in[13];
    const float* Wq2   = (const float*)d_in[14];
    const float* Wk2   = (const float*)d_in[15];
    const float* Wex1  = (const float*)d_in[16];
    const float* bex1  = (const float*)d_in[17];
    const float* Wex2  = (const float*)d_in[18];
    const float* bex2  = (const float*)d_in[19];
    const float* Wo1   = (const float*)d_in[20];
    const float* bo1   = (const float*)d_in[21];
    const float* Wo2   = (const float*)d_in[22];
    const float* bo2   = (const float*)d_in[23];
    const int N = in_sizes[1];
    const int E = in_sizes[2];
    const int T = 3;

    char* wp = (char*)d_ws;
    auto alloc = [&](size_t bytes) -> void* {
        void* r = (void*)wp;
        wp += (bytes + 255) & ~(size_t)255;
        return r;
    };
    float*    Y     = (float*)alloc((size_t)E * 16 * 4);
    float*    Cc    = (float*)alloc((size_t)E * 4);
    ushort_t* RBFb  = (ushort_t*)alloc((size_t)E * 32 * 2);
    ushort_t* wfil  = (ushort_t*)alloc((size_t)E * 128 * 2);
    float*    ev    = (float*)alloc((size_t)N * 16 * 4);
    float*    f     = (float*)alloc((size_t)N * 128 * 4);
    ushort_t* fb    = (ushort_t*)alloc((size_t)N * 128 * 2);
    float*    q     = (float*)alloc((size_t)N * 128 * 4);
    float*    k     = (float*)alloc((size_t)N * 128 * 4);
    float*    v     = (float*)alloc((size_t)N * 128 * 4);
    float*    agg   = (float*)alloc((size_t)N * 128 * 4);
    ushort_t* aggb  = (ushort_t*)alloc((size_t)N * 128 * 2);
    ushort_t* hb    = (ushort_t*)alloc((size_t)N * 160 * 2);
    ushort_t* yb1b  = (ushort_t*)alloc((size_t)N * 128 * 2);
    float*    yb    = (float*)alloc((size_t)N * 144 * 4);
    ushort_t* Wqb   = (ushort_t*)alloc((size_t)T * 128 * 128 * 2);
    ushort_t* Wkb   = (ushort_t*)alloc((size_t)T * 128 * 128 * 2);
    ushort_t* Wvb   = (ushort_t*)alloc((size_t)T * 128 * 128 * 2);
    ushort_t* Wob   = (ushort_t*)alloc((size_t)T * 128 * 128 * 2);
    ushort_t* Wq2b  = (ushort_t*)alloc((size_t)T * 128 * 128 * 2);
    ushort_t* Wk2b  = (ushort_t*)alloc((size_t)T * 128 * 128 * 2);
    ushort_t* Wrbf2b= (ushort_t*)alloc((size_t)T * 128 * 128 * 2);
    ushort_t* Wrbf1b= (ushort_t*)alloc((size_t)T * 128 * 32 * 2);
    ushort_t* Wex1b = (ushort_t*)alloc((size_t)T * 128 * 160 * 2);
    ushort_t* Wex2b = (ushort_t*)alloc((size_t)T * 144 * 128 * 2);
    ushort_t* Wo1b  = (ushort_t*)alloc((size_t)128 * 128 * 2);
    float* hid = q;  // reuse q for the final hidden layer

    auto convW = [&](const float* src, int K, int Nc, ushort_t* dst, int Kp, int Np) {
        int cnt = Np * Kp;
        conv_wT<<<(cnt + 255) / 256, 256, 0, stream>>>(src, K, Nc, dst, Kp, Np);
    };
    for (int t = 0; t < T; t++) {
        convW(Wq + t * 16384, 128, 128, Wqb + t * 16384, 128, 128);
        convW(Wk + t * 16384, 128, 128, Wkb + t * 16384, 128, 128);
        convW(Wv + t * 16384, 128, 128, Wvb + t * 16384, 128, 128);
        convW(Wo + t * 16384, 128, 128, Wob + t * 16384, 128, 128);
        convW(Wq2 + t * 16384, 128, 128, Wq2b + t * 16384, 128, 128);
        convW(Wk2 + t * 16384, 128, 128, Wk2b + t * 16384, 128, 128);
        convW(Wrbf2 + t * 16384, 128, 128, Wrbf2b + t * 16384, 128, 128);
        convW(Wrbf1 + t * 32 * 128, 32, 128, Wrbf1b + t * 128 * 32, 32, 128);
        convW(Wex1 + t * 132 * 128, 132, 128, Wex1b + t * 128 * 160, 160, 128);
        convW(Wex2 + t * 128 * 132, 128, 132, Wex2b + t * 144 * 128, 128, 144);
    }
    convW(Wo1, 128, 128, Wo1b, 128, 128);

    auto gemm = [&](const ushort_t* A, int M, int Kp, const ushort_t* Bt, int Nt,
                    const float* bias, int bl, float* Cf, ushort_t* Cb, int ldc, int mode) {
        dim3 g(((M + 15) / 16 + 7) / 8, Nt);
        switch (mode) {
            case 0: gemm_bf16<false, false, false><<<g, 256, 0, stream>>>(A, M, Kp, Bt, bias, bl, Cf, Cb, ldc); break;
            case 1: gemm_bf16<true,  false, false><<<g, 256, 0, stream>>>(A, M, Kp, Bt, bias, bl, Cf, Cb, ldc); break;
            case 2: gemm_bf16<false, true,  true ><<<g, 256, 0, stream>>>(A, M, Kp, Bt, bias, bl, Cf, Cb, ldc); break;
            case 3: gemm_bf16<false, true,  false><<<g, 256, 0, stream>>>(A, M, Kp, Bt, bias, bl, Cf, Cb, ldc); break;
        }
    };

    // geometry + ev seed + node embedding
    hipMemsetAsync(ev, 0, (size_t)N * 16 * 4, stream);
    edge_geom<<<(E + 255) / 256, 256, 0, stream>>>(positions, senders, receivers, Y, Cc, RBFb, ev, E);
    scale_buf<<<(N * 16 + 255) / 256, 256, 0, stream>>>(ev, N * 16, INV_AVG);
    f_init<<<(N * 128 + 255) / 256, 256, 0, stream>>>(f, embed, species, N);

    int edgeBlocks = ((E + 15) / 16 + 7) / 8;
    int edgeWaveBlocks = (E * 32 + 255) / 256;
    for (int t = 0; t < T; t++) {
        conv_scale<<<(N * 128 + 255) / 256, 256, 0, stream>>>(f, fb, N * 128, 1.0f);
        gemm(fb, N, 128, Wqb + t * 16384, 8, nullptr, 0, q, nullptr, 128, 0);
        gemm(fb, N, 128, Wkb + t * 16384, 8, nullptr, 0, k, nullptr, 128, 0);
        gemm(fb, N, 128, Wvb + t * 16384, 8, nullptr, 0, v, nullptr, 128, 0);
        edge_filter<<<edgeBlocks, 256, 0, stream>>>(RBFb, ev, senders, receivers,
            Wrbf1b + t * 128 * 32, brbf1 + t * 128, Wrbf2b + t * 16384,
            Wdeg + t * 512, wfil, E);
        hipMemsetAsync(agg, 0, (size_t)N * 128 * 4, stream);
        attn_inv<<<edgeWaveBlocks, 256, 0, stream>>>(q, k, v, wfil, Cc, senders, receivers, agg, E);
        conv_scale<<<(N * 128 + 255) / 256, 256, 0, stream>>>(agg, aggb, N * 128, INV_AVG);
        gemm(aggb, N, 128, Wob + t * 16384, 8, nullptr, 0, f, nullptr, 128, 1);   // f += agg@Wo
        conv_scale<<<(N * 128 + 255) / 256, 256, 0, stream>>>(f, fb, N * 128, 1.0f);
        gemm(fb, N, 128, Wq2b + t * 16384, 8, nullptr, 0, q, nullptr, 128, 0);    // q2
        gemm(fb, N, 128, Wk2b + t * 16384, 8, nullptr, 0, k, nullptr, 128, 0);    // k2
        attn_ev<<<edgeWaveBlocks, 256, 0, stream>>>(q, k, wfil, Cc, Y, senders, receivers, ev, E);
        build_h<<<(N * 160 + 255) / 256, 256, 0, stream>>>(f, ev, hb, N);
        gemm(hb, N, 160, Wex1b + t * 128 * 160, 8, bex1 + t * 128, 128, nullptr, yb1b, 128, 2);
        gemm(yb1b, N, 128, Wex2b + t * 144 * 128, 9, bex2 + t * 132, 132, yb, nullptr, 144, 0);
        upd_f<<<(N * 128 + 255) / 256, 256, 0, stream>>>(f, yb, N);
        upd_ev<<<(N * 16 + 255) / 256, 256, 0, stream>>>(ev, yb, N);
    }
    conv_scale<<<(N * 128 + 255) / 256, 256, 0, stream>>>(f, fb, N * 128, 1.0f);
    gemm(fb, N, 128, Wo1b, 8, bo1, 128, hid, nullptr, 128, 3);                    // silu(f@Wo1+bo1)
    hipMemsetAsync(d_out, 0, (size_t)out_size * sizeof(float), stream);
    out_energy<<<(N * 32 + 255) / 256, 256, 0, stream>>>(hid, Wo2, bo2, batch, (float*)d_out, N);
}
